// ResidualBlock1D_43087111913573
// MI455X (gfx1250) — matrix operand order repaired and checked by reading, not compiled, not run on hardware
//
#include <hip/hip_runtime.h>
#include <hip/hip_bf16.h>

// ---------------------------------------------------------------------------
// ResidualBlock1D fused implementation for gfx1250 (MI455X, wave32, WMMA).
//   B=32, Cin=288, Cout=576, Dt=1152, L=4096, G=18.
// bf16 WMMA (v_wmma_f32_16x16x32_bf16) for the three channel GEMMs,
// fp32 accumulation, GroupNorm+SiLU fused into the tile staging, and a
// double-buffered, register-staged K-loop so global loads overlap WMMA.
// A sched_barrier pins all fragment LDS loads ahead of the WMMA quartet so
// the four matrix ops issue back-to-back behind a single s_wait_dscnt.
// ---------------------------------------------------------------------------

typedef __attribute__((ext_vector_type(16))) __bf16 v16bf;
typedef __attribute__((ext_vector_type(8)))  float  v8f;

#define CIN   288
#define COUT  576
#define DT    1152
#define BATCH 32
#define LEN   4096
#define NG    18
#define CPG1  (CIN / NG)    // 16 channels per group, GN1
#define CPG2  (COUT / NG)   // 32 channels per group, GN2
#define EPSV  1e-5f

// GEMM tiling: 64(M) x 128(N) tile per 256-thread block (8 waves of 32).
// Wave grid 4(M) x 2(N): each wave owns 16 rows x 64 cols -> 4 WMMA / K-step.
#define BM    64
#define BN    128
#define BK    32
#define XPAD  40            // bf16 K-stride in LDS: 80B, 16B aligned, spreads banks
#define NT    (LEN / BN)    // 32 N-tiles per batch

#if defined(__has_builtin)
#if __has_builtin(__builtin_amdgcn_sched_barrier)
#define SCHED_FENCE() __builtin_amdgcn_sched_barrier(0)
#endif
#endif
#ifndef SCHED_FENCE
#define SCHED_FENCE()
#endif

// ---------------------------------------------------------------------------
// helpers
// ---------------------------------------------------------------------------
static __device__ __forceinline__ unsigned short bf_bits(float f) {
  union { float f; unsigned u; } v; v.f = f;
  unsigned r = v.u + 0x7FFFu + ((v.u >> 16) & 1u);   // round-to-nearest-even
  return (unsigned short)(r >> 16);
}
static __device__ __forceinline__ float silu(float x) {
  return x / (1.0f + __expf(-x));
}

// A fragment: 16x32 bf16 per documented layout.
// lane = m + 16*half ; elements[0..7] = K half*8.., elements[8..15] = K 16+half*8..
static __device__ __forceinline__ v16bf load_a_frag(const unsigned short* __restrict__ ldsW,
                                                    int wm, int lane) {
  int m = lane & 15, half = lane >> 4;
  const unsigned short* rowp = ldsW + (wm * 16 + m) * XPAD;
  union { v16bf v; uint4 q[2]; } u;
  u.q[0] = *(const uint4*)(rowp + half * 8);
  u.q[1] = *(const uint4*)(rowp + 16 + half * 8);
  return u.v;
}

// B fragment: 32x16 bf16. lane = n + 16*half ; elements[0..7] = K half*8.., elements[8..15] = K 16+half*8..
static __device__ __forceinline__ v16bf load_b_frag(const unsigned short* __restrict__ ldsX,
                                                    int ncol0, int lane) {
  int n = lane & 15, half = lane >> 4;
  const unsigned short* colp = ldsX + (ncol0 + n) * XPAD + half * 8;
  union { v16bf v; uint4 q[2]; } u;
  u.q[0] = *(const uint4*)(colp);
  u.q[1] = *(const uint4*)(colp + 16);
  return u.v;
}

// ---------------------------------------------------------------------------
// Double-buffered, register-staged K-loop shared by both GEMMs.
// MODE 0/1: gn+silu fused into the bf16 convert.  MODE 2: raw bf16 convert.
// One __syncthreads per K-step; next tile's global loads issue right after the
// barrier so they overlap the current step's WMMAs.
// ---------------------------------------------------------------------------
template <int MODE>
static __device__ __forceinline__ void gemm_phase(
    const unsigned short* __restrict__ wbf, int K,
    const float* __restrict__ xsrc, int C,
    int m0, int l0, int b,
    const float* __restrict__ mu, const float* __restrict__ rsig,
    const float* __restrict__ gamma, const float* __restrict__ beta, int cpg,
    unsigned short* __restrict__ ldsW, unsigned short* __restrict__ ldsX,
    v8f (&acc)[4], int wm, int wn, int lane) {
  const int t = threadIdx.x;
  const int wrow = t >> 2, wck = t & 3;          // W tile: 64 rows x 4 x 16B
  const int xj = t & 127, xkb = t >> 7;          // X tile: col j, k = 2*i + xkb
  const int nsteps = K / BK;

  uint4 wreg;
  float xreg[16];

  auto stage = [&](int kk) {                     // gmem -> regs (no LDS)
    wreg = *(const uint4*)(wbf + (size_t)(m0 + wrow) * K + kk + wck * 8);
#pragma unroll
    for (int i = 0; i < 16; ++i) {
      int k = 2 * i + xkb;
      xreg[i] = xsrc[((size_t)b * C + (kk + k)) * LEN + l0 + xj];
    }
  };
  auto commit = [&](int kk, int buf) {           // regs -> LDS (with transform)
    *(uint4*)(ldsW + buf * (BM * XPAD) + wrow * XPAD + wck * 8) = wreg;
    unsigned short* xb = ldsX + buf * (BN * XPAD) + xj * XPAD;
#pragma unroll
    for (int i = 0; i < 16; ++i) {
      int k = 2 * i + xkb;
      int c = kk + k;
      float x = xreg[i], v;
      if (MODE == 2) {
        v = x;
      } else {
        int g = c / cpg;
        float y = (x - mu[b * NG + g]) * rsig[b * NG + g] * gamma[c] + beta[c];
        v = silu(y);
      }
      xb[k] = bf_bits(v);
    }
  };

  stage(0);
  commit(0, 0);
  for (int s = 0; s < nsteps; ++s) {
    __syncthreads();                             // buffer s&1 is ready
    if (s + 1 < nsteps) stage((s + 1) * BK);     // overlap gmem with WMMA
    const unsigned short* Wb = ldsW + (s & 1) * (BM * XPAD);
    const unsigned short* Xb = ldsX + (s & 1) * (BN * XPAD);
    // Load every fragment first; the sched_barrier keeps all five live so the
    // ds_loads clause together and the four WMMAs issue back-to-back.
    v16bf a   = load_a_frag(Wb, wm, lane);
    v16bf bf0 = load_b_frag(Xb, wn * 64 + 0,  lane);
    v16bf bf1 = load_b_frag(Xb, wn * 64 + 16, lane);
    v16bf bf2 = load_b_frag(Xb, wn * 64 + 32, lane);
    v16bf bf3 = load_b_frag(Xb, wn * 64 + 48, lane);
    SCHED_FENCE();
    acc[0] = __builtin_amdgcn_wmma_f32_16x16x32_bf16(false, a, false, bf0, (short)0, acc[0], false, false);
    acc[1] = __builtin_amdgcn_wmma_f32_16x16x32_bf16(false, a, false, bf1, (short)0, acc[1], false, false);
    acc[2] = __builtin_amdgcn_wmma_f32_16x16x32_bf16(false, a, false, bf2, (short)0, acc[2], false, false);
    acc[3] = __builtin_amdgcn_wmma_f32_16x16x32_bf16(false, a, false, bf3, (short)0, acc[3], false, false);
    SCHED_FENCE();
    if (s + 1 < nsteps) commit((s + 1) * BK, (s + 1) & 1);
  }
}

// ---------------------------------------------------------------------------
// kernel 1: convert weight matrices fp32 -> bf16 (one-time, tiny)
// ---------------------------------------------------------------------------
__global__ void k_cvtw(const float* __restrict__ wf, const float* __restrict__ wm,
                       const float* __restrict__ wr,
                       unsigned short* __restrict__ of, unsigned short* __restrict__ om,
                       unsigned short* __restrict__ orr) {
  int i = blockIdx.x * blockDim.x + threadIdx.x;
  if (i < COUT * CIN)  of[i]  = bf_bits(wf[i]);
  if (i < COUT * COUT) om[i]  = bf_bits(wm[i]);
  if (i < COUT * CIN)  orr[i] = bf_bits(wr[i]);
}

// ---------------------------------------------------------------------------
// kernel 2: t[b,o] = silu(time_emb[b,:]) . w_time[o,:] + b_time[o]   (tiny)
// ---------------------------------------------------------------------------
__global__ __launch_bounds__(256) void k_time(const float* __restrict__ temb,
                                              const float* __restrict__ wt,
                                              const float* __restrict__ bt,
                                              float* __restrict__ tvec) {
  int o = blockIdx.x * blockDim.x + threadIdx.x;
  if (o >= BATCH * COUT) return;
  int b = o / COUT, c = o % COUT;
  const float* e = temb + b * DT;
  const float* w = wt + (size_t)c * DT;
  float s = 0.f;
  for (int k = 0; k < DT; ++k) s += silu(e[k]) * w[k];
  tvec[o] = s + bt[c];
}

// ---------------------------------------------------------------------------
// kernel 3: GN1 statistics. one block per (b, g): reduce 16ch x 4096 = 64K vals
// ---------------------------------------------------------------------------
__global__ __launch_bounds__(256) void k_stats1(const float* __restrict__ feature,
                                                float* __restrict__ mu1,
                                                float* __restrict__ rsig1) {
  int b = blockIdx.x / NG, g = blockIdx.x % NG;
  const float4* base4 =
      (const float4*)(feature + ((size_t)b * CIN + g * CPG1) * LEN);
  float s = 0.f, q = 0.f;
  for (int i = threadIdx.x; i < (CPG1 * LEN) / 4; i += 256) {
    float4 x = base4[i];
    s += x.x + x.y + x.z + x.w;
    q += x.x * x.x + x.y * x.y + x.z * x.z + x.w * x.w;
  }
  __shared__ float rs[256], rq[256];
  rs[threadIdx.x] = s; rq[threadIdx.x] = q;
  __syncthreads();
  for (int off = 128; off >= 1; off >>= 1) {
    if (threadIdx.x < off) {
      rs[threadIdx.x] += rs[threadIdx.x + off];
      rq[threadIdx.x] += rq[threadIdx.x + off];
    }
    __syncthreads();
  }
  if (threadIdx.x == 0) {
    float inv = 1.f / (float)(CPG1 * LEN);
    float m = rs[0] * inv;
    float v = rq[0] * inv - m * m;
    mu1[blockIdx.x] = m;
    rsig1[blockIdx.x] = rsqrtf(v + EPSV);
  }
}

// ---------------------------------------------------------------------------
// kernel 4: fused GEMM1.  merged = W_feat x bf16(silu(gn1(feature))) + b + t
// Also emits deterministic per-tile partial (sum, sumsq) for GN2 stats.
// grid: (COUT/BM=9, LEN/BN=32, BATCH=32), 256 threads.
// ---------------------------------------------------------------------------
__global__ __launch_bounds__(256) void k_gemm1(const float* __restrict__ feature,
                                               const unsigned short* __restrict__ wfeat_bf,
                                               const float* __restrict__ b_feat,
                                               const float* __restrict__ tvec,
                                               const float* __restrict__ mu1,
                                               const float* __restrict__ rsig1,
                                               const float* __restrict__ gamma1,
                                               const float* __restrict__ beta1,
                                               float* __restrict__ merged,
                                               float* __restrict__ partS,
                                               float* __restrict__ partQ) {
  const int mt = blockIdx.x, nt = blockIdx.y, b = blockIdx.z;
  const int m0 = mt * BM, l0 = nt * BN;
  const int t = threadIdx.x, lane = t & 31, wid = t >> 5;
  const int wm = wid & 3, wn = wid >> 2;

  __shared__ unsigned short ldsW[2 * BM * XPAD];
  __shared__ unsigned short ldsX[2 * BN * XPAD];

  v8f acc[4];
#pragma unroll
  for (int f = 0; f < 4; ++f) acc[f] = v8f{0.f,0.f,0.f,0.f,0.f,0.f,0.f,0.f};

  gemm_phase<0>(wfeat_bf, CIN, feature, CIN, m0, l0, b,
                mu1, rsig1, gamma1, beta1, CPG1, ldsW, ldsX, acc, wm, wn, lane);

  // epilogue: bias + time-emb add, store merged, accumulate group partials.
  const int n = lane & 15, half = lane >> 4;
  float s = 0.f, q = 0.f;
#pragma unroll
  for (int r = 0; r < 8; ++r) {
    int row = m0 + wm * 16 + half * 8 + r;
    float bias = b_feat[row] + tvec[b * COUT + row];
    size_t base = ((size_t)b * COUT + row) * LEN + l0 + wn * 64;
#pragma unroll
    for (int f = 0; f < 4; ++f) {
      float v = acc[f][r] + bias;
      merged[base + f * 16 + n] = v;
      s += v;
      q += v * v;
    }
  }

  // This 64-row tile spans exactly 2 GN2 groups (32 channels each).
  // Deterministic tree reduction (no float atomics -> bitwise reproducible).
  __shared__ float redS[256], redQ[256];
  int grp = wm >> 1;                                           // 0 or 1
  int idx = grp * 128 + ((wid >> 2) & 1) * 64 + (wm & 1) * 32 + lane;
  redS[idx] = s; redQ[idx] = q;
  __syncthreads();
  for (int off = 64; off >= 1; off >>= 1) {
    if ((idx & 127) < off) {
      redS[idx] += redS[idx + off];
      redQ[idx] += redQ[idx + off];
    }
    __syncthreads();
  }
  if ((idx & 127) == 0) {
    int gg = m0 / 32 + grp;                                    // GN2 group 0..17
    int slot = (b * NG + gg) * NT + nt;
    partS[slot] = redS[idx];
    partQ[slot] = redQ[idx];
  }
}

// ---------------------------------------------------------------------------
// kernel 5: finalize GN2 stats from the 32 partials per (b, g)
// ---------------------------------------------------------------------------
__global__ __launch_bounds__(32) void k_stats2(const float* __restrict__ partS,
                                               const float* __restrict__ partQ,
                                               float* __restrict__ mu2,
                                               float* __restrict__ rsig2) {
  int gslot = blockIdx.x;                                      // 0..575
  __shared__ float rs[32], rq[32];
  int t = threadIdx.x;
  rs[t] = partS[gslot * NT + t];
  rq[t] = partQ[gslot * NT + t];
  __syncthreads();
  for (int off = 16; off >= 1; off >>= 1) {
    if (t < off) { rs[t] += rs[t + off]; rq[t] += rq[t + off]; }
    __syncthreads();
  }
  if (t == 0) {
    float inv = 1.f / (float)(CPG2 * LEN);
    float m = rs[0] * inv;
    float v = rq[0] * inv - m * m;
    mu2[gslot] = m;
    rsig2[gslot] = rsqrtf(v + EPSV);
  }
}

// ---------------------------------------------------------------------------
// kernel 6: fused GEMM2.
//   out = W_merge x bf16(silu(gn2(merged)))  +  W_res x bf16(feature)
//       + b_merge + b_res
// ---------------------------------------------------------------------------
__global__ __launch_bounds__(256) void k_gemm2(const float* __restrict__ merged,
                                               const float* __restrict__ feature,
                                               const unsigned short* __restrict__ wmerge_bf,
                                               const unsigned short* __restrict__ wres_bf,
                                               const float* __restrict__ b_merge,
                                               const float* __restrict__ b_res,
                                               const float* __restrict__ mu2,
                                               const float* __restrict__ rsig2,
                                               const float* __restrict__ gamma2,
                                               const float* __restrict__ beta2,
                                               float* __restrict__ out) {
  const int mt = blockIdx.x, nt = blockIdx.y, b = blockIdx.z;
  const int m0 = mt * BM, l0 = nt * BN;
  const int t = threadIdx.x, lane = t & 31, wid = t >> 5;
  const int wm = wid & 3, wn = wid >> 2;

  __shared__ unsigned short ldsW[2 * BM * XPAD];
  __shared__ unsigned short ldsX[2 * BN * XPAD];

  v8f acc[4];
#pragma unroll
  for (int f = 0; f < 4; ++f) acc[f] = v8f{0.f,0.f,0.f,0.f,0.f,0.f,0.f,0.f};

  // merge path (K = 576), gn2+silu fused into staging.
  gemm_phase<1>(wmerge_bf, COUT, merged, COUT, m0, l0, b,
                mu2, rsig2, gamma2, beta2, CPG2, ldsW, ldsX, acc, wm, wn, lane);
  // residual path (K = 288), raw feature -> bf16.
  gemm_phase<2>(wres_bf, CIN, feature, CIN, m0, l0, b,
                nullptr, nullptr, nullptr, nullptr, CPG1, ldsW, ldsX, acc, wm, wn, lane);

  const int n = lane & 15, half = lane >> 4;
#pragma unroll
  for (int r = 0; r < 8; ++r) {
    int row = m0 + wm * 16 + half * 8 + r;
    float bias = b_merge[row] + b_res[row];
    size_t base = ((size_t)b * COUT + row) * LEN + l0 + wn * 64;
#pragma unroll
    for (int f = 0; f < 4; ++f) {
      out[base + f * 16 + n] = acc[f][r] + bias;
    }
  }
}

// ---------------------------------------------------------------------------
// launcher
// ---------------------------------------------------------------------------
extern "C" void kernel_launch(void* const* d_in, const int* in_sizes, int n_in,
                              void* d_out, int out_size, void* d_ws, size_t ws_size,
                              hipStream_t stream) {
  const float* feature   = (const float*)d_in[0];
  const float* time_emb  = (const float*)d_in[1];
  const float* gn1_gamma = (const float*)d_in[2];
  const float* gn1_beta  = (const float*)d_in[3];
  const float* w_feat    = (const float*)d_in[4];
  const float* b_feat    = (const float*)d_in[5];
  const float* w_time    = (const float*)d_in[6];
  const float* b_time    = (const float*)d_in[7];
  const float* gn2_gamma = (const float*)d_in[8];
  const float* gn2_beta  = (const float*)d_in[9];
  const float* w_merge   = (const float*)d_in[10];
  const float* b_merge   = (const float*)d_in[11];
  const float* w_res     = (const float*)d_in[12];
  const float* b_res     = (const float*)d_in[13];
  float* out = (float*)d_out;

  // workspace carve-up (256B aligned)
  char* ws = (char*)d_ws;
  size_t off = 0;
  auto take = [&](size_t bytes) -> char* {
    char* p = ws + off;
    off = (off + bytes + 255) & ~(size_t)255;
    return p;
  };
  float*          merged    = (float*)         take(sizeof(float) * (size_t)BATCH * COUT * LEN);
  unsigned short* wfeat_bf  = (unsigned short*)take(sizeof(short) * COUT * CIN);
  unsigned short* wmerge_bf = (unsigned short*)take(sizeof(short) * COUT * COUT);
  unsigned short* wres_bf   = (unsigned short*)take(sizeof(short) * COUT * CIN);
  float*          tvec      = (float*)         take(sizeof(float) * BATCH * COUT);
  float*          mu1       = (float*)         take(sizeof(float) * BATCH * NG);
  float*          rsig1     = (float*)         take(sizeof(float) * BATCH * NG);
  float*          mu2       = (float*)         take(sizeof(float) * BATCH * NG);
  float*          rsig2     = (float*)         take(sizeof(float) * BATCH * NG);
  float*          partS     = (float*)         take(sizeof(float) * BATCH * NG * NT);
  float*          partQ     = (float*)         take(sizeof(float) * BATCH * NG * NT);

  // 1) weights -> bf16
  k_cvtw<<<(COUT * COUT + 255) / 256, 256, 0, stream>>>(w_feat, w_merge, w_res,
                                                        wfeat_bf, wmerge_bf, wres_bf);
  // 2) time projection
  k_time<<<(BATCH * COUT + 255) / 256, 256, 0, stream>>>(time_emb, w_time, b_time, tvec);
  // 3) GN1 stats
  k_stats1<<<BATCH * NG, 256, 0, stream>>>(feature, mu1, rsig1);
  // 4) GEMM1 (+ GN2 partials)
  k_gemm1<<<dim3(COUT / BM, LEN / BN, BATCH), 256, 0, stream>>>(
      feature, wfeat_bf, b_feat, tvec, mu1, rsig1, gn1_gamma, gn1_beta,
      merged, partS, partQ);
  // 5) GN2 stats finalize
  k_stats2<<<BATCH * NG, 32, 0, stream>>>(partS, partQ, mu2, rsig2);
  // 6) GEMM2 (merge + residual paths fused)
  k_gemm2<<<dim3(COUT / BM, LEN / BN, BATCH), 256, 0, stream>>>(
      merged, feature, wmerge_bf, wres_bf, b_merge, b_res,
      mu2, rsig2, gn2_gamma, gn2_beta, out);
}
